// NonLocalDenoising_70909910057571
// MI455X (gfx1250) — matrix operand order repaired and checked by reading, not compiled, not run on hardware
//
#include <hip/hip_runtime.h>
#include <hip/hip_bf16.h>
#include <math.h>

// ---------------------------------------------------------------------------
// Problem dims (fixed by the reference)
// ---------------------------------------------------------------------------
#define NB    4
#define C_DIM 256
#define E_DIM 128
#define S_DIM 4096

typedef __bf16 bf16_t;
typedef __attribute__((ext_vector_type(8)))  __bf16 v8bf;
typedef __attribute__((ext_vector_type(16))) __bf16 v16bf;
typedef __attribute__((ext_vector_type(8)))  float  v8f;

// ---------------------------------------------------------------------------
// Fragment builders (CDNA5 WMMA 16-bit layouts, wave32)
//
// A matrix 16x32 (MxK), bf16, 8 VGPRs / 16 elems per lane:
//   lane m = lane&15 ; element j -> K = (j&7) + 8*(lane>>4) + 16*(j>>3)
//   => two contiguous 8-element (16B) loads per fragment.
// B matrix 32x16 (KxN), bf16:
//   lane n = lane&15 ; element j -> K = (j&7) + 8*(lane>>4) + 16*(j>>3)
//   => two contiguous 8-element (16B) loads per fragment.
// C/D 16x16 f32, 8 VGPRs: row M = vr + 8*(lane>>4), col N = lane&15.
// ---------------------------------------------------------------------------
__device__ __forceinline__ v16bf concat16(v8bf lo, v8bf hi) {
  v16bf r;
#pragma unroll
  for (int i = 0; i < 8; ++i) { r[i] = lo[i]; r[i + 8] = hi[i]; }
  return r;
}

__device__ __forceinline__ v16bf make_afrag(const bf16_t* row, int k0, int hi16) {
  v8bf lo = *(const v8bf*)(row + k0 + 8 * hi16);
  v8bf hv = *(const v8bf*)(row + k0 + 16 + 8 * hi16);
  return concat16(lo, hv);
}

__device__ __forceinline__ v16bf make_bfrag(const bf16_t* row, int k0, int hi16) {
  v8bf lo = *(const v8bf*)(row + k0 + 8 * hi16);
  v8bf hv = *(const v8bf*)(row + k0 + 16 + 8 * hi16);
  return concat16(lo, hv);
}

#define WMMA_BF16(a, b, c) \
  __builtin_amdgcn_wmma_f32_16x16x32_bf16(false, (a), false, (b), (short)0, (c), false, false)

// ---------------------------------------------------------------------------
// CDNA5 async global->LDS copy (ASYNCcnt path), GVS addressing:
//   mem = SGPR64 base + VGPR32 byte-offset ; LDS dest byte-offset in vdst VGPR.
// Inline asm per bridge doc (portable across ROCm7.2 / amdgpu-toolchain).
// ---------------------------------------------------------------------------
__device__ __forceinline__ void async_copy_b128(unsigned lds_off,
                                                const void* sbase,
                                                unsigned voff) {
  asm volatile("global_load_async_to_lds_b128 %0, %1, %2"
               :: "v"(lds_off), "v"(voff), "s"(sbase)
               : "memory");
}
__device__ __forceinline__ void wait_async0() {
  asm volatile("s_wait_asynccnt 0x0" ::: "memory");
}
__device__ __forceinline__ unsigned lds_off_u32(const void* p) {
  // shared-aperture flat address: LDS offset lives in the low 32 bits
  return (unsigned)(unsigned long long)p;
}

// ---------------------------------------------------------------------------
// Kernel 1: fp32 -> bf16 conversion (x -> V in [N][C][S] layout, W_proj)
// ---------------------------------------------------------------------------
__global__ void nld_convert(const float* __restrict__ x, const float* __restrict__ Wp,
                            bf16_t* __restrict__ xb, bf16_t* __restrict__ wpb,
                            int nx, int nw) {
  int i = blockIdx.x * blockDim.x + threadIdx.x;
  int stride = gridDim.x * blockDim.x;
  for (int k = i; k < nx; k += stride) xb[k] = (bf16_t)x[k];
  for (int k = i; k < nw; k += stride) wpb[k] = (bf16_t)Wp[k];
}

// ---------------------------------------------------------------------------
// Kernel 2: theta/phi projections.  Q[s][e] = sum_c W_theta[e][c] * x[c][s]
// ---------------------------------------------------------------------------
__global__ void nld_qk(const float* __restrict__ x,
                       const float* __restrict__ Wt,
                       const float* __restrict__ Wf,
                       bf16_t* __restrict__ Qg, bf16_t* __restrict__ Kg) {
  const int e  = threadIdx.x;           // 0..127
  const int s  = blockIdx.x & (S_DIM - 1);
  const int nb = blockIdx.x >> 12;      // S_DIM == 4096 == 2^12
  const float* xc = x + (size_t)nb * C_DIM * S_DIM + s;
  const float* wt = Wt + e * C_DIM;
  const float* wf = Wf + e * C_DIM;
  float at = 0.0f, ap = 0.0f;
#pragma unroll 4
  for (int c = 0; c < C_DIM; ++c) {
    float xv = xc[(size_t)c * S_DIM];
    at = fmaf(wt[c], xv, at);
    ap = fmaf(wf[c], xv, ap);
  }
  size_t o = ((size_t)(nb * S_DIM + s)) * E_DIM + e;
  Qg[o] = (bf16_t)at;
  Kg[o] = (bf16_t)ap;
}

// ---------------------------------------------------------------------------
// Kernel 3: flash attention with double-buffered async K/V tiles.
//   Grid: NB * (S/64) = 256 WGs, 256 threads (8 wave32) each.
// ---------------------------------------------------------------------------
struct __align__(16) AttnSmem {
  bf16_t Qt[64][136];       // [m][e]   (E=128, pad 8)        17408 B
  bf16_t Kt[2][64][136];    // double-buffered [n][e]         34816 B
  bf16_t Vt[2][256][72];    // double-buffered [c][k]         73728 B
  float  Sb[64][65];        // scores                         16640 B
  bf16_t Pb[64][72];        // probabilities (bf16)            9216 B
  float  rmax[64];
  float  rsum[64];
  float  fac[64];
};                           // total ~149 KB -> 2 WGs / 320KB WGP

__launch_bounds__(256, 1)
__global__ void nld_attn(const bf16_t* __restrict__ Xb,   // [N][C][S] (== V^T)
                         const bf16_t* __restrict__ Qg,   // [N][S][E]
                         const bf16_t* __restrict__ Kg,   // [N][S][E]
                         bf16_t* __restrict__ Og) {       // [N][S][C]
  __shared__ AttnSmem sm;
  const int tid  = threadIdx.x;
  const int wave = tid >> 5, lane = tid & 31;
  const int hi16 = lane >> 4, ln = lane & 15;
  const int nb = blockIdx.x >> 6;
  const int s0 = (blockIdx.x & 63) * 64;
  const int mi = wave >> 1, ci = wave & 1;
  const float scale = 0.08838834764831845f;  // 1/sqrt(E=128)

  const bf16_t* Kbase = Kg + (size_t)nb * S_DIM * E_DIM;   // [S][E]
  const bf16_t* Vbase = Xb + (size_t)nb * C_DIM * S_DIM;   // [C][S]

  // ---- issue async K/V fill for tile 0 into buffer 0 ----
  {
#pragma unroll
    for (int i = 0; i < 4; ++i) {
      int idx = tid + i * 256;                 // 1024 chunks of 16B
      int r = idx >> 4, cb = (idx & 15) * 8;
      async_copy_b128(lds_off_u32(&sm.Kt[0][r][cb]), Kbase,
                      (unsigned)((r * E_DIM + cb) * 2));
    }
#pragma unroll
    for (int i = 0; i < 8; ++i) {
      int idx = tid + i * 256;                 // 2048 chunks of 16B
      int r = idx >> 3, cb = (idx & 7) * 8;
      async_copy_b128(lds_off_u32(&sm.Vt[0][r][cb]), Vbase,
                      (unsigned)((r * S_DIM + cb) * 2));
    }
  }

  // ---- load Q tile [64][128] (sync; consumed immediately) ----
  const bf16_t* Qb = Qg + (size_t)(nb * S_DIM + s0) * E_DIM;
  for (int i = tid; i < 64 * 16; i += 256) {
    int r = i >> 4, cb = (i & 15) * 8;
    *(v8bf*)&sm.Qt[r][cb] = *(const v8bf*)(Qb + r * E_DIM + cb);
  }
  if (tid < 64) { sm.rmax[tid] = -3.0e38f; sm.rsum[tid] = 0.0f; }
  __syncthreads();

  // ---- preload this wave's Q fragments (reused for all 64 key tiles) ----
  v16bf qf[4];
  {
    const bf16_t* qrow = &sm.Qt[16 * mi + ln][0];
#pragma unroll
    for (int kc = 0; kc < 4; ++kc) qf[kc] = make_afrag(qrow, 32 * kc, hi16);
  }

  v8f acc[8];
  {
    v8f z = {};
#pragma unroll
    for (int t = 0; t < 8; ++t) acc[t] = z;
  }

  for (int kt = 0; kt < 64; ++kt) {
    const int buf = kt & 1;

    // ---- wait for this tile's async fill, make visible to all waves ----
    wait_async0();
    __syncthreads();

    // ---- issue async prefetch of tile kt+1 into the other buffer ----
    if (kt + 1 < 64) {
      const int nbuf = buf ^ 1;
      const unsigned kboff = (unsigned)((kt + 1) * 64 * E_DIM * 2);  // byte off in Kg
      const unsigned vboff = (unsigned)((kt + 1) * 64 * 2);          // byte off in Xb row dir
#pragma unroll
      for (int i = 0; i < 4; ++i) {
        int idx = tid + i * 256;
        int r = idx >> 4, cb = (idx & 15) * 8;
        async_copy_b128(lds_off_u32(&sm.Kt[nbuf][r][cb]), Kbase,
                        kboff + (unsigned)((r * E_DIM + cb) * 2));
      }
#pragma unroll
      for (int i = 0; i < 8; ++i) {
        int idx = tid + i * 256;
        int r = idx >> 3, cb = (idx & 7) * 8;
        async_copy_b128(lds_off_u32(&sm.Vt[nbuf][r][cb]), Vbase,
                        vboff + (unsigned)((r * S_DIM + cb) * 2));
      }
    }

    // ---- Stage A: S = Q @ K^T (each wave: 2 n-tiles x 4 k-chunks) ----
#pragma unroll
    for (int tt = 0; tt < 2; ++tt) {
      const int nt = (ci << 1) + tt;
      v8f sacc = {};
      const bf16_t* krow = &sm.Kt[buf][16 * nt + ln][0];
#pragma unroll
      for (int kc = 0; kc < 4; ++kc) {
        v16bf b = make_bfrag(krow, 32 * kc, hi16);
        sacc = WMMA_BF16(qf[kc], b, sacc);
      }
#pragma unroll
      for (int vr = 0; vr < 8; ++vr)
        sm.Sb[16 * mi + vr + 8 * hi16][16 * nt + ln] = sacc[vr];
    }
    __syncthreads();

    // ---- Stage B: online softmax (one thread per query row) ----
    if (tid < 64) {
      float mold = sm.rmax[tid];
      float mt = mold;
#pragma unroll 8
      for (int j = 0; j < 64; ++j) mt = fmaxf(mt, sm.Sb[tid][j] * scale);
      float f = __expf(mold - mt);
      float sum = 0.0f;
#pragma unroll 8
      for (int j = 0; j < 64; ++j) {
        float p = __expf(sm.Sb[tid][j] * scale - mt);
        sum += p;
        sm.Pb[tid][j] = (bf16_t)p;
      }
      sm.rsum[tid] = sm.rsum[tid] * f + sum;
      sm.rmax[tid] = mt;
      sm.fac[tid]  = f;
    }
    __syncthreads();

    // ---- Stage C: rescale accumulators, O += P @ V ----
#pragma unroll
    for (int vr = 0; vr < 8; ++vr) {
      float f = sm.fac[16 * mi + vr + 8 * hi16];
#pragma unroll
      for (int t = 0; t < 8; ++t) acc[t][vr] *= f;
    }
    v16bf pf[2];
    {
      const bf16_t* prow = &sm.Pb[16 * mi + ln][0];
      pf[0] = make_afrag(prow, 0, hi16);
      pf[1] = make_afrag(prow, 32, hi16);
    }
#pragma unroll
    for (int t = 0; t < 8; ++t) {
      const bf16_t* vrow = &sm.Vt[buf][128 * ci + 16 * t + ln][0];
      v16bf b0 = make_bfrag(vrow, 0, hi16);
      acc[t] = WMMA_BF16(pf[0], b0, acc[t]);
      v16bf b1 = make_bfrag(vrow, 32, hi16);
      acc[t] = WMMA_BF16(pf[1], b1, acc[t]);
    }
    __syncthreads();   // buffer `buf` free for prefetch issued next iteration
  }

  // ---- finalize: divide by row sums, emit O as bf16 [S][C] ----
#pragma unroll
  for (int vr = 0; vr < 8; ++vr) {
    const int rg = 16 * mi + vr + 8 * hi16;
    const float inv = 1.0f / sm.rsum[rg];
    const size_t srow = (size_t)(nb * S_DIM + s0 + rg) * C_DIM;
#pragma unroll
    for (int t = 0; t < 8; ++t)
      Og[srow + 128 * ci + 16 * t + ln] = (bf16_t)(acc[t][vr] * inv);
  }
}

// ---------------------------------------------------------------------------
// Kernel 4: residual projection.  out[n][o][s] = x[n][o][s] + sum_c Wp[o][c]*O[s][c]
// ---------------------------------------------------------------------------
__launch_bounds__(256, 1)
__global__ void nld_proj(const bf16_t* __restrict__ Og,   // [N][S][C]
                         const bf16_t* __restrict__ Wpb,  // [C][C]
                         const float*  __restrict__ x,
                         float* __restrict__ out) {
  __shared__ __align__(16) bf16_t Ot[64][264];  // [s][c], pad 8
  const int tid  = threadIdx.x;
  const int wave = tid >> 5, lane = tid & 31;
  const int hi16 = lane >> 4, ln = lane & 15;
  const int nb = blockIdx.x >> 6;
  const int s0 = (blockIdx.x & 63) * 64;
  const int mi = wave >> 1, ci = wave & 1;

  const bf16_t* Ob = Og + (size_t)(nb * S_DIM + s0) * C_DIM;
  for (int i = tid; i < 64 * 32; i += 256) {
    int r = i >> 5, cb = (i & 31) * 8;
    *(v8bf*)&Ot[r][cb] = *(const v8bf*)(Ob + r * C_DIM + cb);
  }
  __syncthreads();

  v8f acc[8];
  {
    v8f z = {};
#pragma unroll
    for (int t = 0; t < 8; ++t) acc[t] = z;
  }
  const bf16_t* orow = &Ot[16 * mi + ln][0];
#pragma unroll
  for (int kc = 0; kc < 8; ++kc) {
    v16bf a = make_afrag(orow, 32 * kc, hi16);
#pragma unroll
    for (int t = 0; t < 8; ++t) {
      const int o0 = 128 * ci + 16 * t;
      const bf16_t* wrow = Wpb + (size_t)(o0 + ln) * C_DIM;
      v16bf b = make_bfrag(wrow, 32 * kc, hi16);
      acc[t] = WMMA_BF16(a, b, acc[t]);
    }
  }

  // epilogue: residual add, write to [N][C][S] layout
#pragma unroll
  for (int vr = 0; vr < 8; ++vr) {
    const int srow = s0 + 16 * mi + vr + 8 * hi16;
#pragma unroll
    for (int t = 0; t < 8; ++t) {
      const int o = 128 * ci + 16 * t + ln;
      const size_t idx = ((size_t)nb * C_DIM + o) * S_DIM + srow;
      out[idx] = x[idx] + acc[t][vr];
    }
  }
}

// ---------------------------------------------------------------------------
// Host launcher
// ---------------------------------------------------------------------------
extern "C" void kernel_launch(void* const* d_in, const int* in_sizes, int n_in,
                              void* d_out, int out_size, void* d_ws, size_t ws_size,
                              hipStream_t stream) {
  (void)in_sizes; (void)n_in; (void)out_size; (void)ws_size;
  const float* x  = (const float*)d_in[0];   // [N][C][H][W] = [N][C][S]
  const float* Wt = (const float*)d_in[1];   // [E][C]
  const float* Wf = (const float*)d_in[2];   // [E][C]
  const float* Wp = (const float*)d_in[3];   // [C][C]
  float* out = (float*)d_out;

  const size_t NCS = (size_t)NB * C_DIM * S_DIM;   // 4,194,304
  const size_t NSE = (size_t)NB * S_DIM * E_DIM;   // 2,097,152
  char* ws = (char*)d_ws;
  bf16_t* xb  = (bf16_t*)(ws);                         // 8 MiB
  bf16_t* Qg  = (bf16_t*)(ws + 2 * NCS);               // 4 MiB
  bf16_t* Kg  = (bf16_t*)(ws + 2 * NCS + 2 * NSE);     // 4 MiB
  bf16_t* Og  = (bf16_t*)(ws + 2 * NCS + 4 * NSE);     // 8 MiB
  bf16_t* Wpb = (bf16_t*)(ws + 4 * NCS + 4 * NSE);     // 128 KiB

  nld_convert<<<2048, 256, 0, stream>>>(x, Wp, xb, Wpb, (int)NCS, C_DIM * C_DIM);
  nld_qk<<<NB * S_DIM, E_DIM, 0, stream>>>(x, Wt, Wf, Qg, Kg);
  nld_attn<<<NB * (S_DIM / 64), 256, 0, stream>>>(xb, Qg, Kg, Og);
  nld_proj<<<NB * (S_DIM / 64), 256, 0, stream>>>(Og, Wpb, x, out);
}
